// LSTMLayer_78357383348750
// MI455X (gfx1250) — matrix operand order repaired and checked by reading, not compiled, not run on hardware
//
#include <hip/hip_runtime.h>
#include <hip/hip_bf16.h>
#include <stdint.h>

// Problem dims (from reference setup_inputs)
#define BDIM 64
#define SDIM 256
#define IDIM 512
#define HDIM 1024
#define GDIM (4 * HDIM)  // 4096

typedef __bf16 bf16;
typedef __attribute__((ext_vector_type(8)))  __bf16 v8bf;
typedef __attribute__((ext_vector_type(16))) __bf16 v16bf;
typedef __attribute__((ext_vector_type(8)))  float  v8f;

#if __has_builtin(__builtin_amdgcn_global_load_async_to_lds_b128)
#define USE_ASYNC_LDS 1
#else
#define USE_ASYNC_LDS 0
#endif

// ---------- helpers ----------

// Monotonic float <-> ordered-uint mapping for atomic min/max
__device__ __forceinline__ unsigned fenc(float f) {
    unsigned u = __float_as_uint(f);
    return u ^ ((unsigned)((int)u >> 31) | 0x80000000u);
}
__device__ __forceinline__ float fdec(unsigned u) {
    u = (u & 0x80000000u) ? (u ^ 0x80000000u) : ~u;
    return __uint_as_float(u);
}

// fp32 -> bf16 round-to-nearest-even
__device__ __forceinline__ bf16 f2bf(float f) {
    unsigned u = __float_as_uint(f);
    unsigned short s = (unsigned short)((u + 0x7FFFu + ((u >> 16) & 1u)) >> 16);
    return __builtin_bit_cast(bf16, s);
}

__device__ __forceinline__ float sigmoidf_(float x) {
    return 1.0f / (1.0f + __expf(-x));
}

// Asymmetric signed 8-bit qparams from encoded min/max slots
__device__ __forceinline__ void qparams(unsigned mnE, unsigned mxE, float& scale, float& zp) {
    float mn = fdec(mnE), mx = fdec(mxE);
    scale = fmaxf((mx - mn) * (1.0f / 255.0f), 1e-8f);
    zp = fminf(fmaxf(roundf(-128.0f - mn / scale), -128.0f), 127.0f);
}

#if USE_ASYNC_LDS
typedef int v4i_vs __attribute__((vector_size(16)));
#define AS1Q __attribute__((address_space(1)))
#define AS3Q __attribute__((address_space(3)))

__device__ __forceinline__ void wait_async0() {
#if __has_builtin(__builtin_amdgcn_s_wait_asynccnt)
    __builtin_amdgcn_s_wait_asynccnt(0);
#else
    asm volatile("s_wait_asynccnt 0x0" ::: "memory");
#endif
}

// Per-lane 16B global -> LDS async copy (ASYNCcnt-tracked).
// LDS offset = low 32 bits of flat address.
__device__ __forceinline__ void async_cp16(const void* g, void* l) {
    AS1Q v4i_vs* gp = reinterpret_cast<AS1Q v4i_vs*>((uintptr_t)g);
    AS3Q v4i_vs* lp = reinterpret_cast<AS3Q v4i_vs*>((uint32_t)(uintptr_t)l);
    __builtin_amdgcn_global_load_async_to_lds_b128(gp, lp, 0, 0);
}
#endif

// ---------- tiny control kernels ----------

// slots: even index = running-min (init 0xFFFFFFFF), odd = running-max (init 0)
__global__ void k_reset(unsigned* slots) {
    int i = threadIdx.x;
    if (i < 8) slots[i] = (i & 1) ? 0u : 0xFFFFFFFFu;
}

__global__ __launch_bounds__(256) void k_minmax(const float* __restrict__ x, int n,
                                                unsigned* __restrict__ mnSlot,
                                                unsigned* __restrict__ mxSlot) {
    __shared__ float smn[256], smx[256];
    float mn = 3.4e38f, mx = -3.4e38f;
    for (int i = blockIdx.x * 256 + threadIdx.x; i < n; i += gridDim.x * 256) {
        float v = x[i];
        mn = fminf(mn, v);
        mx = fmaxf(mx, v);
    }
    int t = threadIdx.x;
    smn[t] = mn; smx[t] = mx;
    __syncthreads();
    for (int s = 128; s > 0; s >>= 1) {
        if (t < s) {
            smn[t] = fminf(smn[t], smn[t + s]);
            smx[t] = fmaxf(smx[t], smx[t + s]);
        }
        __syncthreads();
    }
    if (t == 0) {
        atomicMin(mnSlot, fenc(smn[0]));
        atomicMax(mxSlot, fenc(smx[0]));
    }
}

// Quantize the whole input tensor to bf16 integer codes (q - zp); scale_x -> ctrlF[8]
__global__ __launch_bounds__(256) void k_quant_inputs(const float* __restrict__ x,
                                                      bf16* __restrict__ q, int n,
                                                      const unsigned* __restrict__ slots,
                                                      float* __restrict__ ctrlF) {
    float s, zp;
    qparams(slots[0], slots[1], s, zp);
    float inv = 1.0f / s;
    for (int i = blockIdx.x * 256 + threadIdx.x; i < n; i += gridDim.x * 256) {
        float qv = fminf(fmaxf(roundf(x[i] * inv) + zp, -128.0f), 127.0f);
        q[i] = f2bf(qv - zp);
    }
    if (blockIdx.x == 0 && threadIdx.x == 0) ctrlF[8] = s;
}

__global__ __launch_bounds__(256) void k_to_bf16(const float* __restrict__ x,
                                                 bf16* __restrict__ y, int n) {
    for (int i = blockIdx.x * 256 + threadIdx.x; i < n; i += gridDim.x * 256)
        y[i] = f2bf(x[i]);
}

// Final layer-level output fake-quant, in place
__global__ __launch_bounds__(256) void k_requant(float* __restrict__ y, int n,
                                                 const unsigned* __restrict__ slots) {
    float s, zp;
    qparams(slots[0], slots[1], s, zp);
    float inv = 1.0f / s;
    for (int i = blockIdx.x * 256 + threadIdx.x; i < n; i += gridDim.x * 256) {
        float q = fminf(fmaxf(roundf(y[i] * inv) + zp, -128.0f), 127.0f);
        y[i] = (q - zp) * s;
    }
}

// ---------- WMMA gates GEMM ----------
// gates[64,4096] = scale_h * (Hq[64,1024] @ Whh[4096,1024]^T)
//                + scale_x * (Xq_t[64,512] @ Wih[4096,512]^T)
//                + b_ih + b_hh
// Computed as sx * ( (sh/sx) * accH_phase + accX_phase ) + bias with a SINGLE
// accumulator array (rescaled once between phases) to avoid branch-join copies.
// Block = 4 waves covering the FULL M=64 for one 64-column N-strip; the B-operand
// tile (64 cols x 32 K, 4 KB) is identical for all 4 waves, so it is staged once
// per block into double-buffered LDS via async-to-LDS DMA (ASYNCcnt) and consumed
// with ds_load_b128.  W is read exactly once per step (12 MB, L2-resident).
// A frag (16x32 bf16): lane L holds M=L%16; K = kk + (L/16)*16 + e (contiguous 16).
// B frag (32x16 bf16): lane L holds N=L%16; K = kk + (L/16)*16 + e (contiguous 16).
// D frag (16x16 f32):  lane L holds N=L%16; element v -> M = v + (L/16)*8.
#define KSTEPS_H (HDIM / 32)             // 32
#define KSTEPS_X (IDIM / 32)             // 16

__global__ __launch_bounds__(128)
void k_gates_gemm(const bf16* __restrict__ Ah, const bf16* __restrict__ Wh,
                  const bf16* __restrict__ Ax, size_t axRowStride,
                  const bf16* __restrict__ Wx,
                  const float* __restrict__ bih, const float* __restrict__ bhh,
                  const float* __restrict__ ctrlF,
                  float* __restrict__ gates, unsigned* __restrict__ resetSlots) {
    // Reset next-step observer slots (consumed by k_pointwise after this kernel completes)
    if (resetSlots != nullptr && blockIdx.x == 0 && threadIdx.x == 0) {
        resetSlots[0] = 0xFFFFFFFFu; resetSlots[1] = 0u;
        resetSlots[2] = 0xFFFFFFFFu; resetSlots[3] = 0u;
    }

    __shared__ __align__(64) bf16 smem[2][64 * 32];  // double-buffered 4 KB B tiles

    const int wave = threadIdx.x >> 5;   // 0..3 -> M-tile
    const int lane = threadIdx.x & 31;
    const int n0 = blockIdx.x << 6;      // 64-column N strip
    const int lrow = lane & 15;
    const int half = lane >> 4;
    const int b0 = half << 3;

    const bf16* aH = Ah + (size_t)((wave << 4) + lrow) * HDIM;
    const bf16* aX = Ax + (size_t)((wave << 4) + lrow) * axRowStride;

    const v8f vz = {0.f, 0.f, 0.f, 0.f, 0.f, 0.f, 0.f, 0.f};
    v8f acc[4] = {vz, vz, vz, vz};

    // Stage K-step tile `tile` of weight matrix `wsrc` into buffer `buf`:
    // 128 threads x 2 x 16B = 4 KB.
    auto stage = [&](const char* wsrc, size_t rowBytes, int tile, int buf) {
#pragma unroll
        for (int c = 0; c < 2; ++c) {
            int o = (threadIdx.x << 4) + (c << 11);  // byte offset in 4 KB tile
            int row = o >> 6;                        // 0..63 (N within strip)
            int col = o & 63;                        // byte within 64B row segment
            const char* g = wsrc + (size_t)(n0 + row) * rowBytes + (size_t)tile * 64 + col;
            char* l = (char*)&smem[buf][0] + o;
#if USE_ASYNC_LDS
            async_cp16(g, l);
#else
            *(uint4*)l = *(const uint4*)g;           // global_load_b128 + ds_store_b128
#endif
        }
    };

    // Consume the staged tile in buffer `buf` against A fragment at K offset `kk`.
    auto consume = [&](const bf16* ap, int kk, int buf) {
        v8bf alo = *(const v8bf*)(ap + kk + (half << 4));
        v8bf ahi = *(const v8bf*)(ap + kk + (half << 4) + 8);
        v16bf a = __builtin_shufflevector(alo, ahi,
                                          0, 1, 2, 3, 4, 5, 6, 7,
                                          8, 9, 10, 11, 12, 13, 14, 15);
        const bf16* lb = &smem[buf][0];
#pragma unroll
        for (int j = 0; j < 4; ++j) {
            v16bf bm = *(const v16bf*)(lb + (((j << 4) + lrow) << 5) + (half << 4));
            acc[j] = __builtin_amdgcn_wmma_f32_16x16x32_bf16(
                false, a, false, bm, (short)0, acc[j], false, false);
        }
    };

    stage((const char*)Wh, HDIM * 2, 0, 0);
    int cur = 0;

    // ---- phase H: K = 1024 over W_hh ----
#pragma unroll 2
    for (int s = 0; s < KSTEPS_H; ++s) {
#if USE_ASYNC_LDS
        wait_async0();          // this wave's DMA into buf[cur] is complete
#endif
        __syncthreads();        // all waves' DMA visible; prior reads of buf[cur^1] done
        if (s + 1 < KSTEPS_H) stage((const char*)Wh, HDIM * 2, s + 1, cur ^ 1);
        else                  stage((const char*)Wx, IDIM * 2, 0,     cur ^ 1);
        consume(aH, s << 5, cur);
        cur ^= 1;
    }

    // Rescale once: result = sx * ((sh/sx)*accH + accX) + bias
    {
        float r = ctrlF[9] / ctrlF[8];
#pragma unroll
        for (int j = 0; j < 4; ++j)
#pragma unroll
            for (int v = 0; v < 8; ++v) acc[j][v] *= r;
    }

    // ---- phase X: K = 512 over W_ih ----
#pragma unroll 2
    for (int s = 0; s < KSTEPS_X; ++s) {
#if USE_ASYNC_LDS
        wait_async0();
#endif
        __syncthreads();
        if (s + 1 < KSTEPS_X) stage((const char*)Wx, IDIM * 2, s + 1, cur ^ 1);
        consume(aX, s << 5, cur);
        cur ^= 1;
    }

    const float sx = ctrlF[8];
    const int mr = (wave << 4) + (half << 3);
#pragma unroll
    for (int j = 0; j < 4; ++j) {
        int n = n0 + (j << 4) + lrow;
        float bb = bih[n] + bhh[n];
#pragma unroll
        for (int v = 0; v < 8; ++v) {
            gates[(size_t)(mr + v) * GDIM + n] = sx * acc[j][v] + bb;
        }
    }
}

// ---------- per-step pointwise LSTM + min/max observers ----------
__global__ __launch_bounds__(256) void k_pointwise(const float* __restrict__ gates,
                                                   const float* __restrict__ c_dq,
                                                   float* __restrict__ h_raw,
                                                   float* __restrict__ c_raw,
                                                   unsigned* __restrict__ slots) {
    __shared__ float shmn[256], shmx[256], scmn[256], scmx[256];
    int idx = blockIdx.x * 256 + threadIdx.x;  // covers 64*1024 exactly
    int b = idx >> 10;
    int h = idx & (HDIM - 1);
    const float* g = gates + ((size_t)b << 12);
    float ig = sigmoidf_(g[h]);
    float fg = sigmoidf_(g[HDIM + h]);
    float gg = tanhf(g[2 * HDIM + h]);
    float og = sigmoidf_(g[3 * HDIM + h]);
    float cn = fg * c_dq[idx] + ig * gg;
    float hn = og * tanhf(cn);
    h_raw[idx] = hn;
    c_raw[idx] = cn;

    int t = threadIdx.x;
    shmn[t] = hn; shmx[t] = hn; scmn[t] = cn; scmx[t] = cn;
    __syncthreads();
    for (int s = 128; s > 0; s >>= 1) {
        if (t < s) {
            shmn[t] = fminf(shmn[t], shmn[t + s]);
            shmx[t] = fmaxf(shmx[t], shmx[t + s]);
            scmn[t] = fminf(scmn[t], scmn[t + s]);
            scmx[t] = fmaxf(scmx[t], scmx[t + s]);
        }
        __syncthreads();
    }
    if (t == 0) {
        atomicMin(slots + 0, fenc(shmn[0]));
        atomicMax(slots + 1, fenc(shmx[0]));
        atomicMin(slots + 2, fenc(scmn[0]));
        atomicMax(slots + 3, fenc(scmx[0]));
    }
}

// Fake-quant h/c; emit bf16 codes for the next GEMM, dq for outputs; scale_h -> ctrlF[9]
__global__ __launch_bounds__(256) void k_quant_state(
    const float* __restrict__ h_in, const float* __restrict__ c_in,
    bf16* __restrict__ h_q, float* __restrict__ c_dq,
    const unsigned* __restrict__ slots, float* __restrict__ ctrlF,
    float* __restrict__ ys, float* __restrict__ h_last, float* __restrict__ c_last) {
    float hs, hzp, cs, czp;
    qparams(slots[2], slots[3], hs, hzp);
    qparams(slots[4], slots[5], cs, czp);
    int idx = blockIdx.x * 256 + threadIdx.x;  // covers 64*1024 exactly

    float hv = h_in[idx];
    float hq = fminf(fmaxf(roundf(hv / hs) + hzp, -128.0f), 127.0f) - hzp;
    h_q[idx] = f2bf(hq);  // exact: |hq| <= 255 is exactly representable in bf16
    float hdq = hq * hs;
    if (ys) {
        int b = idx >> 10;
        int h = idx & (HDIM - 1);
        ys[(size_t)b * ((size_t)SDIM * HDIM) + h] = hdq;
    }
    if (h_last) h_last[idx] = hdq;

    float cv = c_in[idx];
    float cq = fminf(fmaxf(roundf(cv / cs) + czp, -128.0f), 127.0f) - czp;
    float cdqv = cq * cs;
    c_dq[idx] = cdqv;
    if (c_last) c_last[idx] = cdqv;

    if (blockIdx.x == 0 && threadIdx.x == 0) ctrlF[9] = hs;
}

// ---------- host orchestration ----------
extern "C" void kernel_launch(void* const* d_in, const int* in_sizes, int n_in,
                              void* d_out, int out_size, void* d_ws, size_t ws_size,
                              hipStream_t stream) {
    const float* inputs = (const float*)d_in[0];  // [B,S,I]
    const float* h_prev = (const float*)d_in[1];  // [B,H]
    const float* c_prev = (const float*)d_in[2];  // [B,H]
    const float* W_ih   = (const float*)d_in[3];  // [4H,I]
    const float* W_hh   = (const float*)d_in[4];  // [4H,H]
    const float* b_ih   = (const float*)d_in[5];  // [4H]
    const float* b_hh   = (const float*)d_in[6];  // [4H]
    float* out = (float*)d_out;  // outputs [B,S,H] ++ h_last [B,H] ++ c_last [B,H]

    char* ws = (char*)d_ws;
    size_t off = 0;
    auto alloc = [&](size_t bytes) {
        size_t o = off;
        off += (bytes + 255) & ~(size_t)255;
        return o;
    };
    size_t oCtrl  = alloc(256);
    size_t oWih   = alloc((size_t)GDIM * IDIM * sizeof(bf16));   // 4 MB
    size_t oWhh   = alloc((size_t)GDIM * HDIM * sizeof(bf16));   // 8 MB
    size_t oAq    = alloc((size_t)BDIM * SDIM * IDIM * sizeof(bf16));  // 16 MB
    size_t oGates = alloc((size_t)BDIM * GDIM * sizeof(float));  // 1 MB
    size_t oHraw  = alloc((size_t)BDIM * HDIM * sizeof(float));
    size_t oCraw  = alloc((size_t)BDIM * HDIM * sizeof(float));
    size_t oHq    = alloc((size_t)BDIM * HDIM * sizeof(bf16));
    size_t oCdq   = alloc((size_t)BDIM * HDIM * sizeof(float));
    (void)ws_size;

    unsigned* slots = (unsigned*)(ws + oCtrl);  // 0,1: in  2,3: h  4,5: c  6,7: out
    float* ctrlF    = (float*)(ws + oCtrl);     // [8]=scale_x  [9]=scale_h
    bf16* Wih_bf = (bf16*)(ws + oWih);
    bf16* Whh_bf = (bf16*)(ws + oWhh);
    bf16* Aq     = (bf16*)(ws + oAq);
    float* gates = (float*)(ws + oGates);
    float* Hraw  = (float*)(ws + oHraw);
    float* Craw  = (float*)(ws + oCraw);
    bf16* Hq     = (bf16*)(ws + oHq);
    float* Cdq   = (float*)(ws + oCdq);

    const int nIn = BDIM * SDIM * IDIM;   // 8.4M
    const int nSt = BDIM * HDIM;          // 64K
    const int nOut = BDIM * SDIM * HDIM;  // 16.8M

    // --- prologue: input fake-quant, weight bf16 conversion, initial state quant ---
    k_reset<<<1, 32, 0, stream>>>(slots);
    k_minmax<<<2048, 256, 0, stream>>>(inputs, nIn, slots + 0, slots + 1);
    k_quant_inputs<<<2048, 256, 0, stream>>>(inputs, Aq, nIn, slots, ctrlF);
    k_to_bf16<<<2048, 256, 0, stream>>>(W_ih, Wih_bf, GDIM * IDIM);
    k_to_bf16<<<2048, 256, 0, stream>>>(W_hh, Whh_bf, GDIM * HDIM);
    k_minmax<<<256, 256, 0, stream>>>(h_prev, nSt, slots + 2, slots + 3);
    k_minmax<<<256, 256, 0, stream>>>(c_prev, nSt, slots + 4, slots + 5);
    k_quant_state<<<nSt / 256, 256, 0, stream>>>(h_prev, c_prev, Hq, Cdq, slots, ctrlF,
                                                 nullptr, nullptr, nullptr);

    // --- recurrent scan: 256 steps of WMMA GEMM + pointwise + fake-quant ---
    dim3 gGrid(GDIM / 64);  // 64 blocks, 128 threads = 4 waves sharing one N strip
    for (int t = 0; t < SDIM; ++t) {
        k_gates_gemm<<<gGrid, 128, 0, stream>>>(
            Hq, Whh_bf,
            Aq + (size_t)t * IDIM, (size_t)SDIM * IDIM,
            Wih_bf, b_ih, b_hh, ctrlF, gates, slots + 2);
        k_pointwise<<<nSt / 256, 256, 0, stream>>>(gates, Cdq, Hraw, Craw, slots + 2);
        bool last = (t == SDIM - 1);
        k_quant_state<<<nSt / 256, 256, 0, stream>>>(
            Hraw, Craw, Hq, Cdq, slots, ctrlF,
            out + (size_t)t * HDIM,
            last ? out + (size_t)nOut : nullptr,
            last ? out + (size_t)nOut + (size_t)nSt : nullptr);
    }

    // --- layer-level output fake-quant ---
    k_reset<<<1, 32, 0, stream>>>(slots);
    k_minmax<<<4096, 256, 0, stream>>>(out, nOut, slots + 6, slots + 7);
    k_requant<<<4096, 256, 0, stream>>>(out, nOut, slots + 6);
}
